// MultiHeadAttention_57475252355152
// MI455X (gfx1250) — matrix operand order repaired and checked by reading, not compiled, not run on hardware
//
#include <hip/hip_runtime.h>

// MHA forward for B=2,S=2048,D=1024,H=16,DH=64 on gfx1250 (wave32, WMMA).
// Pipeline: cvt weights -> Q/K/V proj (WMMA f16, unroll-2 reg pipeline) ->
// flash attention (WMMA + double-buffered TDM tensor_load_to_lds of K/V) -> out proj.
// Workspace: Q(8MB) K(8MB) Vt(8MB) O(8MB) Wf16(8MB) = 40MB of d_ws.

#define B_  2
#define S_  2048
#define D_  1024
#define H_  16
#define DH_ 64
#define BS_ (B_ * S_)          // 4096 rows
#define DD_ ((size_t)D_ * D_)  // 1M

typedef __attribute__((ext_vector_type(16))) _Float16 v16h;
typedef __attribute__((ext_vector_type(8)))  _Float16 v8h;
typedef __attribute__((ext_vector_type(8)))  float    v8f;
typedef __attribute__((ext_vector_type(4)))  unsigned int u32x4;
typedef __attribute__((ext_vector_type(8)))  int i32x8;
typedef __attribute__((ext_vector_type(4)))  int i32x4;

#if __has_builtin(__builtin_amdgcn_tensor_load_to_lds)
#define USE_TDM 1
#else
#define USE_TDM 0
#endif

__device__ __forceinline__ v8f v8f_zero() {
  v8f z = {0.f, 0.f, 0.f, 0.f, 0.f, 0.f, 0.f, 0.f};
  return z;
}

__device__ __forceinline__ v8f wmma16(v16h a, v16h b, v8f c) {
  // D = A(16x32 f16) * B(32x16 f16) + C(16x16 f32)
  return __builtin_amdgcn_wmma_f32_16x16x32_f16(
      /*neg_a=*/false, a, /*neg_b=*/false, b,
      /*c_mod=*/(short)0, c, /*reuse_a=*/false, /*reuse_b=*/false);
}

// A-matrix 16x32 f16 fragment. This code's contiguous k order (B matches): lane m=lane&15 is the row,
// half h=lane>>4 selects K half: VGPR v holds K = h*16 + v*2 (+0/+1).
__device__ __forceinline__ v16h load_a16(const _Float16* rowk, int h) {
  v16h a;
#pragma unroll
  for (int v = 0; v < 8; ++v) {
    int k = (h << 4) + (v << 1);
    a[2 * v]     = rowk[k];
    a[2 * v + 1] = rowk[k + 1];
  }
  return a;
}

// Same A pattern, converting from fp32 source on the fly.
__device__ __forceinline__ v16h load_a_f32(const float* rowk, int h) {
  v16h a;
#pragma unroll
  for (int v = 0; v < 8; ++v) {
    int k = (h << 4) + (v << 1);
    float2 f = *(const float2*)(rowk + k);
    a[2 * v]     = (_Float16)f.x;
    a[2 * v + 1] = (_Float16)f.y;
  }
  return a;
}

// B-matrix 32x16 f16 fragment from 32B-aligned global memory.
__device__ __forceinline__ v16h load_b16(const _Float16* p) {
  return *(const v16h*)p;
}

// Same, from 16B-aligned LDS (padded pitch) -> two b128 halves.
__device__ __forceinline__ v16h load_b16_lds(const _Float16* p) {
  v8h lo = *(const v8h*)p;
  v8h hi = *(const v8h*)(p + 8);
  v16h r;
#pragma unroll
  for (int j = 0; j < 8; ++j) { r[j] = lo[j]; r[j + 8] = hi[j]; }
  return r;
}

#define PPITCH 72  // padded LDS row pitch (f16): 64 data + 8 pad = 144B rows

#if USE_TDM
// TDM: DMA one 64-row x 64-elem f16 tile (row-contiguous, dim0_stride elems
// between rows) into LDS with 16B padding after each 128B row -> pitch 72.
// D# per ISA ch.8: group0 {count/type/lds_addr/global_addr}, group1 {dims}.
__device__ __forceinline__ void tdm_load_tile(unsigned lds_addr,
                                              const _Float16* gaddr,
                                              unsigned tensor_d0,
                                              unsigned tensor_d1,
                                              unsigned d0_stride) {
  const unsigned long long ga = (unsigned long long)(size_t)gaddr;
  u32x4 g0;
  g0[0] = 1u;                                     // count=1, user D#
  g0[1] = lds_addr;                               // lds_addr[63:32]
  g0[2] = (unsigned)ga;                           // global_addr[95:64]
  g0[3] = (unsigned)((ga >> 32) & 0x01FFFFFFu) | (2u << 30);  // addr hi | type=2
  i32x8 g1;
  // data_size=1 (2B) | pad_enable | pad_interval=4 (128B) | pad_amount=3 (16B)
  g1[0] = (int)((1u << 16) | (1u << 20) | (4u << 22) | (3u << 25));
  g1[1] = (int)(tensor_d0 << 16);                 // tensor_dim0[15:0] @ bit48
  g1[2] = (int)((tensor_d0 >> 16) | (tensor_d1 << 16));
  g1[3] = (int)((tensor_d1 >> 16) | (64u << 16)); // tile_dim0=64 @ bit112
  g1[4] = 64;                                     // tile_dim1=64, tile_dim2=0
  g1[5] = (int)d0_stride;                         // tensor_dim0_stride[31:0]
  g1[6] = 0;                                      // stride hi | dim1_stride lo
  g1[7] = 0;
  i32x4 z4 = {0, 0, 0, 0};                        // 2D: groups 2/3 unused
  i32x8 z8 = {0, 0, 0, 0, 0, 0, 0, 0};
  __builtin_amdgcn_tensor_load_to_lds(g0, g1, z4, z4, z8, 0);
}
#else
// Fallback: per-lane async DMA, 16 bytes global -> LDS, tracked by ASYNCcnt.
__device__ __forceinline__ void async_copy16(_Float16* ldst, const _Float16* gsrc) {
  unsigned lds = (unsigned)(size_t)ldst;
  asm volatile("global_load_async_to_lds_b128 %0, %1, off"
               :: "v"(lds), "v"(gsrc) : "memory");
}
#endif

// ---------------------------------------------------------------- weights cvt
__global__ void __launch_bounds__(256) cvt_weights_kernel(
    const float* __restrict__ wq, const float* __restrict__ wk,
    const float* __restrict__ wv, const float* __restrict__ wo,
    _Float16* __restrict__ out) {
  size_t stride = (size_t)gridDim.x * blockDim.x;
  for (size_t i = (size_t)blockIdx.x * blockDim.x + threadIdx.x; i < DD_; i += stride) {
    out[i]           = (_Float16)wq[i];
    out[DD_ + i]     = (_Float16)wk[i];
    out[2 * DD_ + i] = (_Float16)wv[i];
    out[3 * DD_ + i] = (_Float16)wo[i];
  }
}

// ------------------------------------------------------- Q/K/V projection
// Y = (X @ W^T + b) * scale, X fp32 (BS x D), W f16 row-major (n,k).
// transposed==0 -> store (B,H,S,DH) ; transposed==1 -> store (B,H,DH,S) (for V).
// K-loop unrolled x2 with two fragment sets: no register-rotation moves.
__global__ void __launch_bounds__(256) proj_qkv_kernel(
    const float* __restrict__ X, const _Float16* __restrict__ W,
    const float* __restrict__ bias, _Float16* __restrict__ out,
    float scale, int transposed) {
  const int lane = threadIdx.x & 31;
  const int wave = threadIdx.x >> 5;
  const int h16  = lane >> 4;
  const int n16  = lane & 15;

  const int tile  = blockIdx.x * 8 + wave;   // 0..4095
  const int mTile = tile >> 4;               // 0..255
  const int nTile = tile & 15;               // 0..15
  const int m0 = mTile << 4;
  const int n0 = nTile << 6;

  const float* arow = X + (size_t)(m0 + n16) * D_;
  const _Float16* wrow[4];
#pragma unroll
  for (int nt = 0; nt < 4; ++nt)
    wrow[nt] = W + (size_t)(n0 + nt * 16 + n16) * D_ + (h16 << 4);

  v8f acc[4];
#pragma unroll
  for (int t = 0; t < 4; ++t) acc[t] = v8f_zero();

  // Set 0 preloaded at k=0; sets alternate every 32 of K.
  v16h a0 = load_a_f32(arow, h16);
  v16h b0[4];
#pragma unroll
  for (int nt = 0; nt < 4; ++nt) b0[nt] = load_b16(wrow[nt]);
  v16h a1, b1[4];

  for (int k0 = 0; k0 < D_ - 64; k0 += 64) {
    a1 = load_a_f32(arow + k0 + 32, h16);
#pragma unroll
    for (int nt = 0; nt < 4; ++nt) b1[nt] = load_b16(wrow[nt] + k0 + 32);
#pragma unroll
    for (int nt = 0; nt < 4; ++nt) acc[nt] = wmma16(a0, b0[nt], acc[nt]);
    a0 = load_a_f32(arow + k0 + 64, h16);
#pragma unroll
    for (int nt = 0; nt < 4; ++nt) b0[nt] = load_b16(wrow[nt] + k0 + 64);
#pragma unroll
    for (int nt = 0; nt < 4; ++nt) acc[nt] = wmma16(a1, b1[nt], acc[nt]);
  }
  // Tail: k = D-64 (set 0) and k = D-32 (set 1).
  a1 = load_a_f32(arow + D_ - 32, h16);
#pragma unroll
  for (int nt = 0; nt < 4; ++nt) b1[nt] = load_b16(wrow[nt] + D_ - 32);
#pragma unroll
  for (int nt = 0; nt < 4; ++nt) acc[nt] = wmma16(a0, b0[nt], acc[nt]);
#pragma unroll
  for (int nt = 0; nt < 4; ++nt) acc[nt] = wmma16(a1, b1[nt], acc[nt]);

  // C layout: lane n=lane&15 is column; VGPR r is row h16*8+r.
#pragma unroll
  for (int nt = 0; nt < 4; ++nt) {
    const int n  = n0 + nt * 16 + n16;
    const float bn = bias[n];
    const int hh = n >> 6, dh = n & 63;
#pragma unroll
    for (int r = 0; r < 8; ++r) {
      const int m = m0 + (h16 << 3) + r;
      const int b = m >> 11, s = m & (S_ - 1);
      const float y = (acc[nt][r] + bn) * scale;
      const size_t idx = transposed
          ? ((((size_t)b * H_ + hh) * DH_ + dh) * S_ + s)
          : ((((size_t)b * H_ + hh) * S_ + s) * DH_ + dh);
      out[idx] = (_Float16)y;
    }
  }
}

// ------------------------------------------------------------ flash attention
// Grid: 512 blocks = (b,h) x (S/128 q-blocks); 8 waves x 16 q-rows each.
// K(B,H,S,DH) and V^T(B,H,DH,S) 64x64 f16 tiles staged into LDS once per block:
// wave0 issues TDM tensor_load_to_lds (TENSORcnt), double-buffered vs compute.
__global__ void __launch_bounds__(256) attn_kernel(
    const _Float16* __restrict__ Qh, const _Float16* __restrict__ Kh,
    const _Float16* __restrict__ Vt, const float* __restrict__ rbias,
    const unsigned char* __restrict__ mask,  // key_padding_mask, 1 byte/bool
    _Float16* __restrict__ Oh) {
  __shared__ __attribute__((aligned(16))) _Float16 Kt[2][64 * PPITCH];   // rows = key
  __shared__ __attribute__((aligned(16))) _Float16 Vs[2][64 * PPITCH];   // rows = dh
  __shared__ __attribute__((aligned(16))) _Float16 Plds[8][16 * PPITCH]; // per-wave P

  const int tid  = threadIdx.x;
  const int lane = tid & 31;
  const int wave = tid >> 5;
  const int h16  = lane >> 4;
  const int n16  = lane & 15;

  const int bid = blockIdx.x;
  const int qb  = bid & 15;          // q block (128 rows)
  const int bh  = bid >> 4;          // b*H + h
  const int b   = bh >> 4;
  const int hd  = bh & 15;
  const int q0  = qb * 128 + wave * 16;

  const _Float16* Qbase = Qh + (size_t)bh * S_ * DH_;
  const _Float16* Kbase = Kh + (size_t)bh * S_ * DH_;
  const _Float16* Vbase = Vt + (size_t)bh * DH_ * S_;
  const float* biasBase = rbias + (size_t)hd * S_ * S_;
  const unsigned char* maskB = mask + (size_t)b * S_;
  _Float16* P = &Plds[wave][0];

  auto stage = [&](int buf, int kb) {
#if USE_TDM
    if (wave == 0) {
      tdm_load_tile((unsigned)(size_t)&Kt[buf][0], Kbase + (size_t)kb * DH_,
                    DH_, S_, DH_);
      tdm_load_tile((unsigned)(size_t)&Vs[buf][0], Vbase + kb,
                    S_, DH_, S_);
    }
#else
#pragma unroll
    for (int c = 0; c < 2; ++c) {
      const int chunk = tid + (c << 8);     // 0..511
      const int row   = chunk >> 3;         // 0..63
      const int col   = (chunk & 7) << 3;   // f16 elements, 16B chunks
      async_copy16(&Kt[buf][row * PPITCH + col],
                   Kbase + (size_t)(kb + row) * DH_ + col);
      async_copy16(&Vs[buf][row * PPITCH + col],
                   Vbase + (size_t)row * S_ + kb + col);
    }
#endif
  };
  auto stage_wait = [&]() {
#if USE_TDM
    if (wave == 0) __builtin_amdgcn_s_wait_tensorcnt(0);
#else
    asm volatile("s_wait_asynccnt 0" ::: "memory");
#endif
    __syncthreads();
  };

  // Q A-fragments (DH=64 -> two 32-wide K chunks), resident for whole kernel.
  const _Float16* qrow = Qbase + (size_t)(q0 + n16) * DH_;
  const v16h qa0 = load_a16(qrow, h16);
  const v16h qa1 = load_a16(qrow + 32, h16);

  v8f o[4];
#pragma unroll
  for (int t = 0; t < 4; ++t) o[t] = v8f_zero();
  float mrun[8], lrun[8];
#pragma unroll
  for (int r = 0; r < 8; ++r) { mrun[r] = -1e30f; lrun[r] = 0.f; }

  stage(0, 0);

  for (int kb = 0, ib = 0; kb < S_; kb += 64, ++ib) {
    const int cur = ib & 1;
    stage_wait();                              // buffer `cur` ready everywhere
    if (kb + 64 < S_) stage(cur ^ 1, kb + 64); // overlap next block's DMA

    const _Float16* Ktile = &Kt[cur][0];
    const _Float16* Vtile = &Vs[cur][0];

    // S = Q K^T (+bias, mask): 8 WMMAs -> 4 f32 16x16 tiles.
    float p[4][8];
#pragma unroll
    for (int nt = 0; nt < 4; ++nt) {
      const int key = kb + nt * 16 + n16;
      const _Float16* krow = Ktile + (nt * 16 + n16) * PPITCH + (h16 << 4);
      v8f c = v8f_zero();
      c = wmma16(qa0, load_b16_lds(krow), c);
      c = wmma16(qa1, load_b16_lds(krow + 32), c);
      const bool valid = maskB[key] != 0;
      const float* bp = biasBase + (size_t)(q0 + (h16 << 3)) * S_ + key;
#pragma unroll
      for (int r = 0; r < 8; ++r)
        p[nt][r] = valid ? (c[r] + bp[(size_t)r * S_]) : -1e30f;
    }

    // Online softmax: row = (h16*8 + r); reduce across the 16 lanes of a half.
#pragma unroll
    for (int r = 0; r < 8; ++r) {
      float t = fmaxf(fmaxf(p[0][r], p[1][r]), fmaxf(p[2][r], p[3][r]));
#pragma unroll
      for (int off = 8; off >= 1; off >>= 1)
        t = fmaxf(t, __shfl_xor(t, off, 16));
      const float mnew  = fmaxf(mrun[r], t);
      const float alpha = __expf(mrun[r] - mnew);
      mrun[r] = mnew;
      float rs = 0.f;
#pragma unroll
      for (int nt = 0; nt < 4; ++nt) {
        const float e = __expf(p[nt][r] - mnew);
        p[nt][r] = e;
        rs += e;
      }
#pragma unroll
      for (int off = 8; off >= 1; off >>= 1)
        rs += __shfl_xor(rs, off, 16);
      lrun[r] = lrun[r] * alpha + rs;
#pragma unroll
      for (int nt = 0; nt < 4; ++nt) o[nt][r] *= alpha;
    }

    // Stage P (16x64 f16) in wave-private LDS, reload in A layout.
#pragma unroll
    for (int nt = 0; nt < 4; ++nt)
#pragma unroll
      for (int r = 0; r < 8; ++r)
        P[((h16 << 3) + r) * PPITCH + nt * 16 + n16] = (_Float16)p[nt][r];
    asm volatile("s_wait_dscnt 0" ::: "memory");

    const v16h pa0 = load_a16(P + n16 * PPITCH, h16);
    const v16h pa1 = load_a16(P + n16 * PPITCH + 32, h16);

    // O += P * V : 8 WMMAs with native C accumulation.
#pragma unroll
    for (int nt = 0; nt < 4; ++nt) {
      const _Float16* vp = Vtile + (nt * 16 + n16) * PPITCH + (h16 << 4);
      o[nt] = wmma16(pa0, load_b16_lds(vp), o[nt]);
      o[nt] = wmma16(pa1, load_b16_lds(vp + 32), o[nt]);
    }
  }

  // Normalize and store to (B,S,D) f16.
#pragma unroll
  for (int r = 0; r < 8; ++r) {
    const float inv = 1.0f / lrun[r];
    const int s = q0 + (h16 << 3) + r;
#pragma unroll
    for (int nt = 0; nt < 4; ++nt) {
      const int dh = nt * 16 + n16;
      Oh[((size_t)b * S_ + s) * D_ + hd * DH_ + dh] = (_Float16)(o[nt][r] * inv);
    }
  }
}

// --------------------------------------------------------- output projection
// out(fp32) = Oh(f16) @ Wo^T + bo, unroll-2 pipeline like proj_qkv.
__global__ void __launch_bounds__(256) proj_out_kernel(
    const _Float16* __restrict__ Xh, const _Float16* __restrict__ W,
    const float* __restrict__ bias, float* __restrict__ out) {
  const int lane = threadIdx.x & 31;
  const int wave = threadIdx.x >> 5;
  const int h16  = lane >> 4;
  const int n16  = lane & 15;

  const int tile  = blockIdx.x * 8 + wave;
  const int mTile = tile >> 4;
  const int nTile = tile & 15;
  const int m0 = mTile << 4;
  const int n0 = nTile << 6;

  const _Float16* arow = Xh + (size_t)(m0 + n16) * D_;
  const _Float16* wrow[4];
#pragma unroll
  for (int nt = 0; nt < 4; ++nt)
    wrow[nt] = W + (size_t)(n0 + nt * 16 + n16) * D_ + (h16 << 4);

  v8f acc[4];
#pragma unroll
  for (int t = 0; t < 4; ++t) acc[t] = v8f_zero();

  v16h a0 = load_a16(arow, h16);
  v16h b0[4];
#pragma unroll
  for (int nt = 0; nt < 4; ++nt) b0[nt] = load_b16(wrow[nt]);
  v16h a1, b1[4];

  for (int k0 = 0; k0 < D_ - 64; k0 += 64) {
    a1 = load_a16(arow + k0 + 32, h16);
#pragma unroll
    for (int nt = 0; nt < 4; ++nt) b1[nt] = load_b16(wrow[nt] + k0 + 32);
#pragma unroll
    for (int nt = 0; nt < 4; ++nt) acc[nt] = wmma16(a0, b0[nt], acc[nt]);
    a0 = load_a16(arow + k0 + 64, h16);
#pragma unroll
    for (int nt = 0; nt < 4; ++nt) b0[nt] = load_b16(wrow[nt] + k0 + 64);
#pragma unroll
    for (int nt = 0; nt < 4; ++nt) acc[nt] = wmma16(a1, b1[nt], acc[nt]);
  }
  a1 = load_a16(arow + D_ - 32, h16);
#pragma unroll
  for (int nt = 0; nt < 4; ++nt) b1[nt] = load_b16(wrow[nt] + D_ - 32);
#pragma unroll
  for (int nt = 0; nt < 4; ++nt) acc[nt] = wmma16(a0, b0[nt], acc[nt]);
#pragma unroll
  for (int nt = 0; nt < 4; ++nt) acc[nt] = wmma16(a1, b1[nt], acc[nt]);

#pragma unroll
  for (int nt = 0; nt < 4; ++nt) {
    const int n = n0 + nt * 16 + n16;
    const float bn = bias[n];
#pragma unroll
    for (int r = 0; r < 8; ++r) {
      const int m = m0 + (h16 << 3) + r;
      out[(size_t)m * D_ + n] = acc[nt][r] + bn;
    }
  }
}

// ---------------------------------------------------------------------- launch
extern "C" void kernel_launch(void* const* d_in, const int* in_sizes, int n_in,
                              void* d_out, int out_size, void* d_ws, size_t ws_size,
                              hipStream_t stream) {
  const float* query = (const float*)d_in[0];
  const float* key   = (const float*)d_in[1];
  const float* value = (const float*)d_in[2];
  const unsigned char* mask = (const unsigned char*)d_in[3];  // jnp bool -> 1 byte
  const float* rbias = (const float*)d_in[4];
  const float* Wq = (const float*)d_in[5];   const float* bq = (const float*)d_in[6];
  const float* Wk = (const float*)d_in[7];   const float* bk = (const float*)d_in[8];
  const float* Wv = (const float*)d_in[9];   const float* bv = (const float*)d_in[10];
  const float* Wo = (const float*)d_in[11];  const float* bo = (const float*)d_in[12];

  _Float16* ws = (_Float16*)d_ws;
  const size_t QKV = (size_t)B_ * H_ * S_ * DH_;  // 4M elements
  _Float16* Qh = ws;
  _Float16* Kh = Qh + QKV;
  _Float16* Vh = Kh + QKV;                        // transposed (B,H,DH,S)
  _Float16* Oh = Vh + QKV;                        // (B,S,D)
  _Float16* Wh = Oh + (size_t)BS_ * D_;           // 4 x (D*D) f16: q,k,v,o

  cvt_weights_kernel<<<512, 256, 0, stream>>>(Wq, Wk, Wv, Wo, Wh);

  const float qscale = 0.125f;  // 1/sqrt(DH), exact in f16
  proj_qkv_kernel<<<512, 256, 0, stream>>>(query, Wh,            bq, Qh, qscale, 0);
  proj_qkv_kernel<<<512, 256, 0, stream>>>(key,   Wh + DD_,      bk, Kh, 1.0f,   0);
  proj_qkv_kernel<<<512, 256, 0, stream>>>(value, Wh + 2 * DD_,  bv, Vh, 1.0f,   1);

  attn_kernel<<<512, 256, 0, stream>>>(Qh, Kh, Vh, rbias, mask, Oh);

  proj_out_kernel<<<512, 256, 0, stream>>>(Oh, Wh + 3 * DD_, bo, (float*)d_out);
}
